// CausalSelfAttention_36825049596076
// MI455X (gfx1250) — matrix operand order repaired and checked by reading, not compiled, not run on hardware
//
#include <hip/hip_runtime.h>
#include <hip/hip_bf16.h>

typedef __bf16 bf16;
typedef __bf16 v16bf __attribute__((ext_vector_type(16)));
typedef __bf16 v8bf  __attribute__((ext_vector_type(8)));
typedef float  v8f   __attribute__((ext_vector_type(8)));
typedef float  v4f   __attribute__((ext_vector_type(4)));

constexpr int Bsz = 2, Tseq = 2048, Dmodel = 1024, Hn = 16, HDim = 64;

// ---- fragment helpers -------------------------------------------------------

// A fragment from bf16 memory (global or LDS): contiguous [p0, p0+16)
__device__ __forceinline__ v16bf load_a_bf16(const bf16* __restrict__ p0) {
  v8bf lo = *(const v8bf*)(p0);
  v8bf hh = *(const v8bf*)(p0 + 8);
  return __builtin_shufflevector(lo, hh, 0, 1, 2, 3, 4, 5, 6, 7, 8, 9, 10, 11, 12, 13, 14, 15);
}

__device__ __forceinline__ v8f wmma_bf16(v16bf a, v16bf b, v8f c) {
  return __builtin_amdgcn_wmma_f32_16x16x32_bf16(false, a, false, b, (short)0, c, false, false);
}

// ---- kernel 0: fp32 -> bf16 bulk convert (bandwidth-bound, one pass) -------
__global__ __launch_bounds__(256) void cvt_bf16_kernel(const float* __restrict__ src,
                                                       bf16* __restrict__ dst, int n8) {
  const int i = blockIdx.x * 256 + threadIdx.x;
  if (i < n8) {
    v4f f0 = *(const v4f*)(src + (size_t)i * 8);
    v4f f1 = *(const v4f*)(src + (size_t)i * 8 + 4);
    v8bf r;
#pragma unroll
    for (int j = 0; j < 4; ++j) {
      r[j] = (bf16)f0[j];
      r[j + 4] = (bf16)f1[j];
    }
    *(v8bf*)(dst + (size_t)i * 8) = r;
  }
}

// ---- kernel 1: fused QKV projection + bias + RoPE --------------------------
// C[m, n] = sum_k Xb[m,k] * Wb[n,k] + b[n];  n in [0, 3072).  All bf16 operands.
// K-loop fully unrolled: every load is base + immediate offset (no address
// arithmetic in the hot loop); backend pipelines the 160-load/128-WMMA window.
__global__ __launch_bounds__(256) void qkv_rope_kernel(
    const bf16* __restrict__ Xb, const bf16* __restrict__ Wb,
    const float* __restrict__ cosT, const float* __restrict__ sinT,
    const float* __restrict__ bq, const float* __restrict__ bk,
    const float* __restrict__ bv,
    bf16* __restrict__ Qo, bf16* __restrict__ Ko, bf16* __restrict__ VTo) {
  const int lane = threadIdx.x & 31;
  const int wave = threadIdx.x >> 5;
  const int m0 = (blockIdx.x * 8 + wave) * 16;
  const int ncol0 = blockIdx.y * 64;   // global column across [Wq; Wk; Wv]
  const int mat = ncol0 >> 10;         // 0=Q, 1=K, 2=V
  const int n0 = ncol0 & 1023;
  const float* __restrict__ bias = (mat == 0) ? bq : ((mat == 1) ? bk : bv);

  const int row = lane & 15;
  const int hi  = lane >> 4;

  const bf16* __restrict__ xrow = Xb + (size_t)(m0 + row) * Dmodel + hi * 16;
  const bf16* __restrict__ wrow[4];
#pragma unroll
  for (int t = 0; t < 4; ++t)
    wrow[t] = Wb + (size_t)(ncol0 + t * 16 + row) * Dmodel + hi * 16;

  v8f acc[4] = {};
#pragma unroll
  for (int k0 = 0; k0 < Dmodel; k0 += 32) {
    v16bf a = load_a_bf16(xrow + k0);
    v16bf bfr[4];
#pragma unroll
    for (int t = 0; t < 4; ++t) bfr[t] = *(const v16bf*)(wrow[t] + k0);
#pragma unroll
    for (int t = 0; t < 4; ++t) acc[t] = wmma_bf16(a, bfr[t], acc[t]);
  }

#pragma unroll
  for (int t = 0; t < 4; ++t) {
    const int n = n0 + t * 16 + row;
    const float bn = bias[n];
    const int h  = n >> 6;
    const int hd = n & 63;
#pragma unroll
    for (int i = 0; i < 8; ++i) {
      const int m = m0 + i + 8 * hi;
      const int bidx = m >> 11;
      const int tp = m & (Tseq - 1);
      float v = acc[t][i] + bn;
      if (mat < 2) {
        const int pi = hd >> 1;
        const float c = cosT[tp * (HDim / 2) + pi];
        const float s = sinT[tp * (HDim / 2) + pi];
        const float pv = __shfl_xor(v, 1, 32);  // RoPE pair partner (hd^1)
        float o = (hd & 1) ? (pv * s + v * c) : (v * c - pv * s);
        if (mat == 0) o *= 0.125f;  // 1/sqrt(64) folded into Q
        bf16* dst = (mat == 0) ? Qo : Ko;
        dst[((size_t)(bidx * Hn + h) * Tseq + tp) * HDim + hd] = (bf16)o;
      } else {
        VTo[((size_t)(bidx * Hn + h) * HDim + hd) * Tseq + tp] = (bf16)v;
      }
    }
  }
}

// ---- kernel 2: causal flash attention --------------------------------------
__global__ __launch_bounds__(256) void flash_attn_kernel(
    const bf16* __restrict__ Q, const bf16* __restrict__ K,
    const bf16* __restrict__ VT, bf16* __restrict__ O) {
  __shared__ __align__(32) bf16 Plds[8][16 * 64];  // wave-private P tiles
  const int lane = threadIdx.x & 31;
  const int wave = threadIdx.x >> 5;
  const int bh = blockIdx.x;
  const int b = bh >> 4, h = bh & 15;
  const int q0 = blockIdx.y * 128 + wave * 16;
  const int row = lane & 15, hi = lane >> 4;

  const bf16* __restrict__ Qb = Q + (size_t)bh * Tseq * HDim;
  const bf16* __restrict__ Kb = K + (size_t)bh * Tseq * HDim;
  const bf16* __restrict__ Vb = VT + (size_t)bh * HDim * Tseq;

  v16bf qa[2];
  {
    const bf16* qrow = Qb + (size_t)(q0 + row) * HDim;
#pragma unroll
    for (int ks = 0; ks < 2; ++ks) qa[ks] = load_a_bf16(qrow + ks * 32 + hi * 16);
  }

  v8f o[4] = {};
  float mrow[8], lrow[8];
#pragma unroll
  for (int i = 0; i < 8; ++i) { mrow[i] = -3.0e38f; lrow[i] = 0.f; }

  const int ktmax = (q0 + 15) >> 6;
  for (int kt = 0; kt <= ktmax; ++kt) {
    const int kb = kt * 64;
    if (kt < ktmax) {
      __builtin_prefetch(Kb + (size_t)(kb + 64 + row) * HDim, 0, 1);
      __builtin_prefetch(Vb + (size_t)row * Tseq + kb + 64, 0, 1);
    }

    // Load all 8 K fragments up front (one load clause), then 8 WMMAs.
    v16bf kf[2][4];
#pragma unroll
    for (int ks = 0; ks < 2; ++ks)
#pragma unroll
      for (int t = 0; t < 4; ++t)
        kf[ks][t] = *(const v16bf*)(Kb + (size_t)(kb + t * 16 + row) * HDim +
                                    ks * 32 + hi * 16);
    v8f s[4] = {};
#pragma unroll
    for (int ks = 0; ks < 2; ++ks)
#pragma unroll
      for (int t = 0; t < 4; ++t) s[t] = wmma_bf16(qa[ks], kf[ks][t], s[t]);

    // Issue V fragment loads now; softmax VALU work below hides their latency.
    v16bf vf[2][4];
#pragma unroll
    for (int ks = 0; ks < 2; ++ks)
#pragma unroll
      for (int t = 0; t < 4; ++t)
        vf[ks][t] = *(const v16bf*)(Vb + (size_t)(t * 16 + row) * Tseq + kb +
                                    ks * 32 + hi * 16);

    // causal mask on diagonal tiles
    if (kb + 63 > q0) {
#pragma unroll
      for (int t = 0; t < 4; ++t) {
        const int kg = kb + t * 16 + row;
#pragma unroll
        for (int i = 0; i < 8; ++i)
          if (kg > q0 + i + 8 * hi) s[t][i] = -3.0e38f;
      }
    }

    // online softmax; rows live across lanes {0..15}/{16..31}
#pragma unroll
    for (int i = 0; i < 8; ++i) {
      float mx = fmaxf(fmaxf(s[0][i], s[1][i]), fmaxf(s[2][i], s[3][i]));
#pragma unroll
      for (int d = 1; d < 16; d <<= 1) mx = fmaxf(mx, __shfl_xor(mx, d, 32));
      const float mn = fmaxf(mrow[i], mx);
      const float alpha = __expf(mrow[i] - mn);
      mrow[i] = mn;
      float rs = 0.f;
#pragma unroll
      for (int t = 0; t < 4; ++t) {
        const float e = __expf(s[t][i] - mn);
        s[t][i] = e;
        rs += e;
      }
#pragma unroll
      for (int d = 1; d < 16; d <<= 1) rs += __shfl_xor(rs, d, 32);
      lrow[i] = lrow[i] * alpha + rs;
#pragma unroll
      for (int t = 0; t < 4; ++t) o[t][i] *= alpha;
    }

    // P: C-layout -> A-layout through wave-private LDS tile
    bf16* __restrict__ pl = &Plds[wave][0];
#pragma unroll
    for (int t = 0; t < 4; ++t)
#pragma unroll
      for (int i = 0; i < 8; ++i)
        pl[(i + 8 * hi) * 64 + t * 16 + row] = (bf16)s[t][i];

#pragma unroll
    for (int ks = 0; ks < 2; ++ks) {
      v16bf pa = load_a_bf16(pl + row * 64 + ks * 32 + hi * 16);
#pragma unroll
      for (int t = 0; t < 4; ++t) o[t] = wmma_bf16(pa, vf[ks][t], o[t]);
    }
  }

#pragma unroll
  for (int i = 0; i < 8; ++i) {
    const float inv = 1.0f / lrow[i];
    const int qg = q0 + i + 8 * hi;
#pragma unroll
    for (int t = 0; t < 4; ++t)
      O[(size_t)(b * Tseq + qg) * Dmodel + h * HDim + t * 16 + row] = (bf16)(o[t][i] * inv);
  }
}

// ---- kernel 3: output projection  Y = O . Wo^T + bo ------------------------
__global__ __launch_bounds__(256) void oproj_kernel(
    const bf16* __restrict__ A, const bf16* __restrict__ Wob,
    const float* __restrict__ bo, float* __restrict__ Y) {
  const int lane = threadIdx.x & 31;
  const int wave = threadIdx.x >> 5;
  const int m0 = (blockIdx.x * 8 + wave) * 16;
  const int n0 = blockIdx.y * 64;
  const int row = lane & 15, hi = lane >> 4;

  const bf16* __restrict__ arow = A + (size_t)(m0 + row) * Dmodel + hi * 16;
  const bf16* __restrict__ wrow[4];
#pragma unroll
  for (int t = 0; t < 4; ++t)
    wrow[t] = Wob + (size_t)(n0 + t * 16 + row) * Dmodel + hi * 16;

  v8f acc[4] = {};
#pragma unroll
  for (int k0 = 0; k0 < Dmodel; k0 += 32) {
    v16bf a = load_a_bf16(arow + k0);
    v16bf bfr[4];
#pragma unroll
    for (int t = 0; t < 4; ++t) bfr[t] = *(const v16bf*)(wrow[t] + k0);
#pragma unroll
    for (int t = 0; t < 4; ++t) acc[t] = wmma_bf16(a, bfr[t], acc[t]);
  }

#pragma unroll
  for (int t = 0; t < 4; ++t) {
    const int n = n0 + t * 16 + row;
    const float bn = bo[n];
#pragma unroll
    for (int i = 0; i < 8; ++i)
      Y[(size_t)(m0 + i + 8 * hi) * Dmodel + n] = acc[t][i] + bn;
  }
}

// ---- launcher ---------------------------------------------------------------
extern "C" void kernel_launch(void* const* d_in, const int* in_sizes, int n_in,
                              void* d_out, int out_size, void* d_ws, size_t ws_size,
                              hipStream_t stream) {
  const float* x    = (const float*)d_in[0];
  const float* cosT = (const float*)d_in[1];
  const float* sinT = (const float*)d_in[2];
  const float* Wq = (const float*)d_in[3];
  const float* bq = (const float*)d_in[4];
  const float* Wk = (const float*)d_in[5];
  const float* bk = (const float*)d_in[6];
  const float* Wv = (const float*)d_in[7];
  const float* bv = (const float*)d_in[8];
  const float* Wo = (const float*)d_in[9];
  const float* bo = (const float*)d_in[10];

  const size_t nX   = (size_t)Bsz * Tseq * Dmodel;       // 4,194,304
  const size_t nW   = (size_t)Dmodel * Dmodel;           // 1,048,576
  const size_t nQKV = (size_t)Bsz * Hn * Tseq * HDim;    // 4,194,304

  bf16* Xb    = (bf16*)d_ws;      // x as bf16                 [B*T, D]
  bf16* Wqkvb = Xb + nX;          // [Wq; Wk; Wv] as bf16      [3D, D]
  bf16* Wob   = Wqkvb + 3 * nW;   // Wo as bf16                [D, D]
  bf16* Qw    = Wob + nW;         // RoPE'd, pre-scaled Q      [B,H,T,HD]
  bf16* Kw    = Qw + nQKV;        // RoPE'd K                  [B,H,T,HD]
  bf16* Vw    = Kw + nQKV;        // V transposed              [B,H,HD,T]
  bf16* Ow    = Vw + nQKV;        // attention output          [B*T, D]

  // one-pass fp32 -> bf16 conversions
  cvt_bf16_kernel<<<dim3((int)(nX / 8 / 256)), 256, 0, stream>>>(x, Xb, (int)(nX / 8));
  cvt_bf16_kernel<<<dim3((int)(nW / 8 / 256)), 256, 0, stream>>>(Wq, Wqkvb, (int)(nW / 8));
  cvt_bf16_kernel<<<dim3((int)(nW / 8 / 256)), 256, 0, stream>>>(Wk, Wqkvb + nW, (int)(nW / 8));
  cvt_bf16_kernel<<<dim3((int)(nW / 8 / 256)), 256, 0, stream>>>(Wv, Wqkvb + 2 * nW, (int)(nW / 8));
  cvt_bf16_kernel<<<dim3((int)(nW / 8 / 256)), 256, 0, stream>>>(Wo, Wob, (int)(nW / 8));

  qkv_rope_kernel<<<dim3(32, 48), 256, 0, stream>>>(Xb, Wqkvb, cosT, sinT, bq, bk, bv,
                                                    Qw, Kw, Vw);
  flash_attn_kernel<<<dim3(32, 16), 256, 0, stream>>>(Qw, Kw, Vw, Ow);
  oproj_kernel<<<dim3(32, 16), 256, 0, stream>>>(Ow, Wob, bo, (float*)d_out);
}
